// GraphTransformerEncoderLayer_7842610282648
// MI455X (gfx1250) — matrix operand order repaired and checked by reading, not compiled, not run on hardware
//
#include <hip/hip_runtime.h>

// ---------------------------------------------------------------------------
// Types for CDNA5 WMMA (wave32, v_wmma_f32_16x16x32_f16)
// ---------------------------------------------------------------------------
typedef _Float16 f16x8  __attribute__((ext_vector_type(8)));
typedef _Float16 f16x16 __attribute__((ext_vector_type(16)));
typedef float    f32x8  __attribute__((ext_vector_type(8)));

__device__ __forceinline__ f16x16 cat8(f16x8 lo, f16x8 hi) {
    return __builtin_shufflevector(lo, hi, 0,1,2,3,4,5,6,7,8,9,10,11,12,13,14,15);
}

__device__ __forceinline__ f32x8 wmma_f16(f16x16 a, f16x16 b, f32x8 c) {
    // D = A(16x32) * B(32x16) + C(16x16), f32 accumulate
    return __builtin_amdgcn_wmma_f32_16x16x32_f16(false, a, false, b, (short)0, c,
                                                  false, false);
}

#define GN 4096
#define GC 1024
#define GH 4
#define GD 256
#define GFF 4096

// ---------------------------------------------------------------------------
// Elementwise f32 -> f16 cast
// ---------------------------------------------------------------------------
__global__ __launch_bounds__(256) void castf16(const float* __restrict__ in,
                                               _Float16* __restrict__ out, int n) {
    for (int i = blockIdx.x * 256 + threadIdx.x; i < n; i += gridDim.x * 256)
        out[i] = (_Float16)in[i];
}

__global__ __launch_bounds__(256) void fzero(float* __restrict__ p, size_t n) {
    size_t i = (size_t)blockIdx.x * 256 + threadIdx.x;
    if (i < n) p[i] = 0.0f;
}

// ---------------------------------------------------------------------------
// Row squared-norms of an f16 [rows, 256] matrix; one wave per row.
// ---------------------------------------------------------------------------
__global__ __launch_bounds__(256) void rownorm2(const _Float16* __restrict__ X,
                                                float* __restrict__ out, int rows) {
    const int lane = threadIdx.x & 31, wid = threadIdx.x >> 5;
    const int row = blockIdx.x * 8 + wid;
    if (row >= rows) return;
    f16x8 v = *(const f16x8*)(X + (size_t)row * GD + lane * 8);
    float s = 0.0f;
    for (int e = 0; e < 8; ++e) { float x = (float)v[e]; s += x * x; }
    for (int off = 16; off; off >>= 1) s += __shfl_xor(s, off, 32);
    if (lane == 0) out[row] = s;
}

// ---------------------------------------------------------------------------
// Dense adjacency scatter with faithful view(1,2,N,N) channel-slab conv fold:
//   flat([N,N,2]) index f = i*2N + 2j + c ; slab0 (f < N*N) gets conv_w[0],
//   slab1 gets conv_w[1]; result accumulated into adjc[N*N].
// ---------------------------------------------------------------------------
__global__ __launch_bounds__(256) void scatter_adj(const int* __restrict__ ei,
                                                   const float* __restrict__ sgn,
                                                   const float* __restrict__ conv_w,
                                                   float* __restrict__ adjc,
                                                   int E, int Nn) {
    int e = blockIdx.x * 256 + threadIdx.x;
    if (e >= E) return;
    long long i = ei[e], j = ei[E + e];
    float w0 = conv_w[0], w1 = conv_w[1];
    long long NN = (long long)Nn * Nn;
    long long f = i * (2LL * Nn) + 2LL * j;
    for (int c = 0; c < 2; ++c) {
        long long fc = f + c;
        float wgt = (fc < NN) ? w0 : w1;
        long long g = (fc < NN) ? fc : fc - NN;
        atomicAdd(&adjc[g], wgt * sgn[2 * e + c]);
    }
}

// ---------------------------------------------------------------------------
// Generic f16 WMMA GEMM:  out[M,Nn] = act(A[M,K] @ W[Nn,K]^T + bias)
// Block: 256 threads (8 waves), 128x256 tile. Wave grid 2x4, wave tile 64x64:
// 16 WMMAs per 8 LDS fragment loads per K-step.
// ---------------------------------------------------------------------------
#define LDA 40  // LDS row stride in halves (pad 32 -> 40 against bank conflicts)

__global__ __launch_bounds__(256) void gemm_f16(const _Float16* __restrict__ A,
                                                const _Float16* __restrict__ W,
                                                const float* __restrict__ bias,
                                                float* __restrict__ Cf,
                                                _Float16* __restrict__ Ch,
                                                int M, int Nn, int K, int relu) {
    __shared__ _Float16 As[128 * LDA];
    __shared__ _Float16 Ws[256 * LDA];
    const int tid = threadIdx.x;
    const int lane = tid & 31, w = tid >> 5;
    const int wm = (w & 1) * 64, wn = (w >> 1) * 64;
    const int m0 = blockIdx.y * 128, n0 = blockIdx.x * 256;
    const int lm = lane & 15, hs = lane >> 4;  // lane-half select

    f32x8 acc[4][4];
    for (int i = 0; i < 4; ++i)
        for (int j = 0; j < 4; ++j) acc[i][j] = {};

    for (int k0 = 0; k0 < K; k0 += 32) {
        // cooperative tile load: A 128x32 halves, W 256x32 halves
        for (int ld = 0; ld < 2; ++ld) {
            int id = tid + ld * 256;
            int r = id >> 2, c8 = (id & 3) << 3;
            *(f16x8*)&As[r * LDA + c8] =
                *(const f16x8*)&A[(size_t)(m0 + r) * K + k0 + c8];
        }
        for (int ld = 0; ld < 4; ++ld) {
            int id = tid + ld * 256;
            int r = id >> 2, c8 = (id & 3) << 3;
            *(f16x8*)&Ws[r * LDA + c8] =
                *(const f16x8*)&W[(size_t)(n0 + r) * K + k0 + c8];
        }
        if (k0 + 32 < K) {  // prefetch next tile into cache hierarchy
            __builtin_prefetch(&A[(size_t)(m0 + (tid >> 1)) * K + k0 + 32], 0, 1);
            __builtin_prefetch(&W[(size_t)(n0 + (tid >> 1)) * K + k0 + 32], 0, 1);
            __builtin_prefetch(&W[(size_t)(n0 + 128 + (tid >> 1)) * K + k0 + 32], 0, 1);
        }
        __syncthreads();

        // A fragments: lanes 0-15 -> K 0..15, lanes 16-31 -> K 16..31
        f16x16 af[4];
        for (int i = 0; i < 4; ++i) {
            const _Float16* p = &As[(wm + i * 16 + lm) * LDA + hs * 16];
            af[i] = cat8(*(const f16x8*)p, *(const f16x8*)(p + 8));
        }
        // B fragments: lanes 0-15 -> K 0..15, lanes 16-31 -> K 16..31
        f16x16 bf[4];
        for (int j = 0; j < 4; ++j) {
            const _Float16* p = &Ws[(wn + j * 16 + lm) * LDA + hs * 16];
            bf[j] = cat8(*(const f16x8*)p, *(const f16x8*)(p + 8));
        }
        for (int i = 0; i < 4; ++i)
            for (int j = 0; j < 4; ++j) acc[i][j] = wmma_f16(af[i], bf[j], acc[i][j]);
        __syncthreads();
    }

    // Epilogue: C/D layout -> VGPR r holds row r (lanes<16) / r+8 (lanes>=16)
    for (int j = 0; j < 4; ++j) {
        int col = n0 + wn + j * 16 + lm;
        float bv = bias ? bias[col] : 0.0f;
        for (int i = 0; i < 4; ++i) {
            for (int r = 0; r < 8; ++r) {
                int row = m0 + wm + i * 16 + r + hs * 8;
                float v = acc[i][j][r] + bv;
                if (relu) v = fmaxf(v, 0.0f);
                size_t idx = (size_t)row * Nn + col;
                if (Cf) Cf[idx] = v;
                if (Ch) Ch[idx] = (_Float16)v;
            }
        }
    }
}

// ---------------------------------------------------------------------------
// Flash attention with L2-distance logits + adjacency bias, per head.
// 4 waves/block; each wave owns one 16-row query block, streams 32-key tiles.
// q2/k2 row norms are precomputed (rownorm2) and read as scalars.
// ---------------------------------------------------------------------------
__global__ __launch_bounds__(128) void flash_attn(const _Float16* __restrict__ Q,
                                                  const _Float16* __restrict__ Km,
                                                  const _Float16* __restrict__ V,
                                                  const float* __restrict__ adjc,
                                                  const float* __restrict__ q2buf,
                                                  const float* __restrict__ k2buf,
                                                  const float* __restrict__ param_p,
                                                  const float* __restrict__ convb_p,
                                                  _Float16* __restrict__ O,
                                                  int Nn, int Dd) {
    __shared__ _Float16 Pl[4][16 * 32];    // P tile, 16 q x 32 k, per wave
    __shared__ _Float16 Vt[4][128 * 32];   // V half-tile transposed [d][key]

    const int tid = threadIdx.x, lane = tid & 31, wid = tid >> 5;
    const int nblk = Nn / 16;
    const int gw = blockIdx.x * 4 + wid;
    const int h = gw / nblk, qb = gw % nblk;
    const size_t hoff = (size_t)h * Nn * Dd;
    const _Float16* Qh = Q + hoff;
    const _Float16* Kh = Km + hoff;
    const _Float16* Vh = V + hoff;
    const int lm = lane & 15, hs = lane >> 4;
    const int qrow0 = qb * 16;

    // Q fragments (A-layout) for the whole D=256 (8 chunks of K=32)
    f16x16 qa[8];
    for (int c = 0; c < 8; ++c) {
        const _Float16* p = Qh + (size_t)(qrow0 + lm) * Dd + c * 32 + hs * 16;
        qa[c] = cat8(*(const f16x8*)p, *(const f16x8*)(p + 8));
    }
    float q2r[8];
    for (int r = 0; r < 8; ++r) q2r[r] = q2buf[h * Nn + qrow0 + r + hs * 8];

    const float pm = *param_p, cb = *convb_p;
    const float invs = 0.03125f;  // 1/sqrt(C) = 1/32

    float mrun[8], lrun[8];
    for (int r = 0; r < 8; ++r) { mrun[r] = -3.0e38f; lrun[r] = 0.0f; }
    f32x8 oacc[16];
    for (int t = 0; t < 16; ++t) oacc[t] = {};

    for (int kt = 0; kt < Nn / 32; ++kt) {
        const int kbase = kt * 32;
        // S = q . k^T over D, two 16-key column tiles
        f32x8 S0 = {}, S1 = {};
        for (int c = 0; c < 8; ++c) {
            const _Float16* p0 = Kh + (size_t)(kbase + lm) * Dd + c * 32 + hs * 16;
            const _Float16* p1 = Kh + (size_t)(kbase + 16 + lm) * Dd + c * 32 + hs * 16;
            f16x16 b0 = cat8(*(const f16x8*)p0, *(const f16x8*)(p0 + 8));
            f16x16 b1 = cat8(*(const f16x8*)p1, *(const f16x8*)(p1 + 8));
            S0 = wmma_f16(qa[c], b0, S0);
            S1 = wmma_f16(qa[c], b1, S1);
        }
        float k2a = k2buf[h * Nn + kbase + lm];
        float k2b = k2buf[h * Nn + kbase + 16 + lm];

        // logits + online softmax (row r lives on VGPR r, lane-half selects +8)
        for (int r = 0; r < 8; ++r) {
            int row = qrow0 + r + hs * 8;
            int c0 = kbase + lm;
            float l0 = (q2r[r] + k2a - 2.0f * S0[r]) * invs +
                       pm * (adjc[(size_t)row * Nn + c0] + cb);
            float l1 = (q2r[r] + k2b - 2.0f * S1[r]) * invs +
                       pm * (adjc[(size_t)row * Nn + c0 + 16] + cb);
            float mx = fmaxf(l0, l1);
            for (int off = 8; off; off >>= 1) mx = fmaxf(mx, __shfl_xor(mx, off, 32));
            float mn = fmaxf(mrun[r], mx);
            float scl = __expf(mrun[r] - mn);
            float p0 = __expf(l0 - mn), p1 = __expf(l1 - mn);
            float rs = p0 + p1;
            for (int off = 8; off; off >>= 1) rs += __shfl_xor(rs, off, 32);
            lrun[r] = lrun[r] * scl + rs;
            mrun[r] = mn;
            for (int t = 0; t < 16; ++t) oacc[t][r] *= scl;
            Pl[wid][(r + hs * 8) * 32 + lm] = (_Float16)p0;
            Pl[wid][(r + hs * 8) * 32 + 16 + lm] = (_Float16)p1;
        }
        __syncthreads();
        asm volatile("s_wait_dscnt 0" ::: "memory");

        // P A-fragment from LDS (rows lm, split K halves per lane group)
        f16x16 pa;
        {
            const _Float16* pp = &Pl[wid][lm * 32 + hs * 16];
            pa = cat8(*(const f16x8*)pp, *(const f16x8*)(pp + 8));
        }

        // O += P @ V, V staged transposed in LDS, 128-d halves
        for (int dh = 0; dh < 2; ++dh) {
            const _Float16* vp = Vh + (size_t)(kbase + lane) * Dd + dh * 128;
            for (int db = 0; db < 16; ++db) {
                f16x8 vv = *(const f16x8*)(vp + db * 8);
                for (int jj = 0; jj < 8; ++jj)
                    Vt[wid][(db * 8 + jj) * 32 + lane] = vv[jj];
            }
            __syncthreads();
            asm volatile("s_wait_dscnt 0" ::: "memory");
            for (int t = 0; t < 8; ++t) {
                const _Float16* vq = &Vt[wid][(t * 16 + lm) * 32 + hs * 16];
                f16x16 vb = cat8(*(const f16x8*)vq, *(const f16x8*)(vq + 8));
                oacc[dh * 8 + t] = wmma_f16(pa, vb, oacc[dh * 8 + t]);
            }
            __syncthreads();
        }
    }

    // normalize and store f16 (flat reshape(N,C) == contiguous head slabs)
    for (int t = 0; t < 16; ++t) {
        for (int r = 0; r < 8; ++r) {
            int row = qrow0 + r + hs * 8;
            int d = t * 16 + lm;
            float v = oacc[t][r] / lrun[r];
            O[hoff + (size_t)row * Dd + d] = (_Float16)v;
        }
    }
}

// ---------------------------------------------------------------------------
// Residual add + LayerNorm (+ optional per-channel PReLU); one row per block.
// ---------------------------------------------------------------------------
__global__ __launch_bounds__(256) void ln_kernel(const float* __restrict__ A,
                                                 const float* __restrict__ B,
                                                 const float* __restrict__ g,
                                                 const float* __restrict__ bt,
                                                 const float* __restrict__ prelu,
                                                 float* __restrict__ outf,
                                                 _Float16* __restrict__ outh,
                                                 int Cdim) {
    const int row = blockIdx.x, tid = threadIdx.x;
    const float* a = A + (size_t)row * Cdim;
    const float* b = B + (size_t)row * Cdim;
    float v[4], s = 0.0f, ss = 0.0f;
    for (int i = 0; i < 4; ++i) {
        int c = tid + i * 256;
        float x = a[c] + b[c];
        v[i] = x; s += x; ss += x * x;
    }
    for (int off = 16; off; off >>= 1) {
        s += __shfl_xor(s, off, 32);
        ss += __shfl_xor(ss, off, 32);
    }
    __shared__ float ps[8], pss[8];
    if ((tid & 31) == 0) { ps[tid >> 5] = s; pss[tid >> 5] = ss; }
    __syncthreads();
    if (tid == 0) {
        float t = 0.0f, tt = 0.0f;
        for (int i = 0; i < 8; ++i) { t += ps[i]; tt += pss[i]; }
        ps[0] = t; pss[0] = tt;
    }
    __syncthreads();
    float mu = ps[0] / Cdim;
    float var = pss[0] / Cdim - mu * mu;
    float rs = rsqrtf(var + 1e-5f);
    for (int i = 0; i < 4; ++i) {
        int c = tid + i * 256;
        float y = (v[i] - mu) * rs * g[c] + bt[c];
        if (prelu) y = (y >= 0.0f) ? y : prelu[c] * y;
        if (outf) outf[(size_t)row * Cdim + c] = y;
        if (outh) outh[(size_t)row * Cdim + c] = (_Float16)y;
    }
}

// ---------------------------------------------------------------------------
// Host orchestration
// ---------------------------------------------------------------------------
extern "C" void kernel_launch(void* const* d_in, const int* in_sizes, int n_in,
                              void* d_out, int out_size, void* d_ws, size_t ws_size,
                              hipStream_t stream) {
    (void)n_in; (void)out_size; (void)ws_size;
    const float* x     = (const float*)d_in[0];
    const int*   ei    = (const int*)d_in[1];
    const float* sgn   = (const float*)d_in[2];
    const float* wq    = (const float*)d_in[3];
    const float* wk    = (const float*)d_in[4];
    const float* wv    = (const float*)d_in[5];
    const float* wo    = (const float*)d_in[6];
    const float* convw = (const float*)d_in[7];
    const float* convb = (const float*)d_in[8];
    const float* param = (const float*)d_in[9];
    const float* ln1g  = (const float*)d_in[10];
    const float* ln1b  = (const float*)d_in[11];
    const float* ff1w  = (const float*)d_in[12];
    const float* ff1b  = (const float*)d_in[13];
    const float* ff2w  = (const float*)d_in[14];
    const float* ff2b  = (const float*)d_in[15];
    const float* ln2g  = (const float*)d_in[16];
    const float* ln2b  = (const float*)d_in[17];
    const float* prelu = (const float*)d_in[18];

    const int N = GN, C = GC, D = GD, FF = GFF, H = GH;
    const int E = in_sizes[1] / 2;

    char* ws = (char*)d_ws;
    size_t o = 0;
    auto take = [&](size_t bytes) -> char* {
        char* p = ws + o;
        o += (bytes + 255) & ~(size_t)255;
        return p;
    };
    _Float16* xb    = (_Float16*)take((size_t)N * C * 2);
    _Float16* wqh   = (_Float16*)take((size_t)C * C * 2);
    _Float16* wkh   = (_Float16*)take((size_t)C * C * 2);
    _Float16* wvh   = (_Float16*)take((size_t)C * C * 2);
    _Float16* woh   = (_Float16*)take((size_t)C * C * 2);
    _Float16* ff1h  = (_Float16*)take((size_t)FF * C * 2);
    _Float16* ff2h  = (_Float16*)take((size_t)C * FF * 2);
    _Float16* qh_   = (_Float16*)take((size_t)N * C * 2);
    _Float16* kh_   = (_Float16*)take((size_t)N * C * 2);
    _Float16* vh_   = (_Float16*)take((size_t)N * C * 2);
    float*    q2b   = (float*)take((size_t)H * N * 4);
    float*    k2b   = (float*)take((size_t)H * N * 4);
    float*    adjc  = (float*)take((size_t)N * N * 4);
    _Float16* attnh = (_Float16*)take((size_t)N * C * 2);
    float*    attno = (float*)take((size_t)N * C * 4);
    float*    out1f = (float*)take((size_t)N * C * 4);
    _Float16* out1h = (_Float16*)take((size_t)N * C * 2);
    _Float16* ffhh  = (_Float16*)take((size_t)N * FF * 2);
    float*    ff2o  = (float*)take((size_t)N * C * 4);

    auto cast = [&](const float* src, _Float16* dst, int n) {
        castf16<<<(n + 255) / 256, 256, 0, stream>>>(src, dst, n);
    };
    cast(x, xb, N * C);
    cast(wq, wqh, C * C);
    cast(wk, wkh, C * C);
    cast(wv, wvh, C * C);
    cast(wo, woh, C * C);
    cast(ff1w, ff1h, FF * C);
    cast(ff2w, ff2h, C * FF);

    dim3 blk(256);
    dim3 gCC(C / 256, N / 128);
    gemm_f16<<<gCC, blk, 0, stream>>>(xb, wqh, nullptr, nullptr, qh_, N, C, C, 0);
    gemm_f16<<<gCC, blk, 0, stream>>>(xb, wkh, nullptr, nullptr, kh_, N, C, C, 0);
    gemm_f16<<<gCC, blk, 0, stream>>>(xb, wvh, nullptr, nullptr, vh_, N, C, C, 0);

    // per-row |q|^2, |k|^2 over the [H*N, D] head view
    rownorm2<<<(H * N) / 8, 256, 0, stream>>>(qh_, q2b, H * N);
    rownorm2<<<(H * N) / 8, 256, 0, stream>>>(kh_, k2b, H * N);

    size_t NN = (size_t)N * N;
    fzero<<<(unsigned)((NN + 255) / 256), 256, 0, stream>>>(adjc, NN);
    scatter_adj<<<(E + 255) / 256, 256, 0, stream>>>(ei, sgn, convw, adjc, E, N);

    flash_attn<<<H * (N / 16) / 4, 128, 0, stream>>>(qh_, kh_, vh_, adjc, q2b, k2b,
                                                     param, convb, attnh, N, D);

    gemm_f16<<<gCC, blk, 0, stream>>>(attnh, woh, nullptr, attno, nullptr, N, C, C, 0);
    ln_kernel<<<N, 256, 0, stream>>>(attno, x, ln1g, ln1b, nullptr, out1f, out1h, C);

    dim3 gFF(FF / 256, N / 128);
    gemm_f16<<<gFF, blk, 0, stream>>>(out1h, ff1h, ff1b, nullptr, ffhh, N, FF, C, 1);
    gemm_f16<<<gCC, blk, 0, stream>>>(ffhh, ff2h, ff2b, ff2o, nullptr, N, C, FF, 0);

    ln_kernel<<<N, 256, 0, stream>>>(ff2o, out1f, ln2g, ln2b, prelu,
                                     (float*)d_out, nullptr, C);
}
